// TransformerRLPolicy_21526376087639
// MI455X (gfx1250) — matrix operand order repaired and checked by reading, not compiled, not run on hardware
//
#include <hip/hip_runtime.h>
#include <hip/hip_fp16.h>

typedef __attribute__((ext_vector_type(16))) _Float16 v16h;
typedef __attribute__((ext_vector_type(8)))  _Float16 v8h;
typedef __attribute__((ext_vector_type(8)))  float    v8f;

#define BN   4096
#define SEQ  53
#define DIM  256
#define NH   8
#define DH   32
#define NL   6
#define FF   1024
#define MAXB 50
#define MTOT ((size_t)BN * SEQ)   // 217088, multiple of 32

// ---------------------------------------------------------------------------
// CDNA5 helpers
// ---------------------------------------------------------------------------
static __device__ __forceinline__ v16h ldfrag(const _Float16* lo, const _Float16* hi) {
    v8h a = *(const v8h*)lo;
    v8h b = *(const v8h*)hi;
    v16h r;
#pragma unroll
    for (int i = 0; i < 8; ++i) { r[i] = a[i]; r[i + 8] = b[i]; }
    return r;
}

static __device__ __forceinline__ v8f wmma_f16(v16h a, v16h b, v8f c) {
    return __builtin_amdgcn_wmma_f32_16x16x32_f16(
        /*neg_a=*/false, a, /*neg_b=*/false, b,
        /*c_mod=*/(short)0, c, /*reuse_a=*/false, /*reuse_b=*/false);
}

// Async global->LDS copy of 16 bytes per lane (CDNA5 GLOBAL_LOAD_ASYNC_TO_LDS,
// tracked with ASYNCcnt). lds_off = low 32 bits of the generic shared pointer.
static __device__ __forceinline__ void async_copy_b128(uint32_t lds_off, const void* gaddr) {
    asm volatile("global_load_async_to_lds_b128 %0, %1, off"
                 :
                 : "v"(lds_off), "v"(gaddr)
                 : "memory");
}
static __device__ __forceinline__ void async_wait0() {
    asm volatile("s_wait_asynccnt 0x0" ::: "memory");
}
static __device__ __forceinline__ uint32_t lds_off32(const void* p) {
    return (uint32_t)(uintptr_t)p;
}

// ---------------------------------------------------------------------------
// Weight convert: dst[n*K + k] = (f16) src[k*N + n]
// ---------------------------------------------------------------------------
__global__ void convert_wT(const float* __restrict__ src, _Float16* __restrict__ dst,
                           int K, int N) {
    int idx = blockIdx.x * 256 + threadIdx.x;
    if (idx < K * N) {
        int n = idx / K, k = idx % K;
        dst[idx] = (_Float16)src[(size_t)k * N + n];
    }
}

// ---------------------------------------------------------------------------
// Embedding + boid packing. One block per batch row.
// ---------------------------------------------------------------------------
__global__ __launch_bounds__(256) void embed_kernel(
    const float* __restrict__ obs,
    const float* __restrict__ cls_emb,  const float* __restrict__ type_cls,
    const float* __restrict__ type_ctx, const float* __restrict__ type_pred,
    const float* __restrict__ type_boid,
    const float* __restrict__ ctxW, const float* __restrict__ ctxb,
    const float* __restrict__ predW, const float* __restrict__ predb,
    const float* __restrict__ boidW, const float* __restrict__ boidb,
    float* __restrict__ x32, _Float16* __restrict__ x16, int* __restrict__ nvalid)
{
    const int b = blockIdx.x, tid = threadIdx.x;
    __shared__ float ob[4 + 4 * MAXB];
    __shared__ int ord[MAXB];
    __shared__ int nv;
    for (int i = tid; i < 4 + 4 * MAXB; i += 256) ob[i] = obs[(size_t)b * (4 + 4 * MAXB) + i];
    __syncthreads();
    if (tid == 0) {
        int c = 0;
        for (int j = 0; j < MAXB; ++j) {
            const float* p = &ob[4 + 4 * j];
            if (p[0] != 0.f || p[1] != 0.f || p[2] != 0.f || p[3] != 0.f) ord[c++] = j;
        }
        nv = c;
        for (int j = 0; j < MAXB; ++j) {
            const float* p = &ob[4 + 4 * j];
            if (!(p[0] != 0.f || p[1] != 0.f || p[2] != 0.f || p[3] != 0.f)) ord[c++] = j;
        }
        nvalid[b] = nv;
    }
    __syncthreads();
    const int d = tid;
    const size_t base = (size_t)b * SEQ * DIM;
    float t0 = cls_emb[d] + type_cls[d];
    x32[base + d] = t0; x16[base + d] = (_Float16)t0;
    float t1 = ob[0] * ctxW[d] + ob[1] * ctxW[DIM + d] + ctxb[d] + type_ctx[d];
    x32[base + DIM + d] = t1; x16[base + DIM + d] = (_Float16)t1;
    float t2 = ob[2] * predW[d] + ob[3] * predW[DIM + d] + predb[d] + type_pred[d];
    x32[base + 2 * DIM + d] = t2; x16[base + 2 * DIM + d] = (_Float16)t2;
    const int nvl = nv;
    for (int j = 0; j < MAXB; ++j) {
        float t = 0.f;
        int sj = ord[j];
        if (j < nvl) {
            const float* p = &ob[4 + 4 * sj];
            t = p[0] * boidW[d] + p[1] * boidW[DIM + d] + p[2] * boidW[2 * DIM + d]
              + p[3] * boidW[3 * DIM + d] + boidb[d] + type_boid[d];
        }
        x32[base + (size_t)(3 + j) * DIM + d] = t;
        x16[base + (size_t)(3 + j) * DIM + d] = (_Float16)t;
    }
}

// ---------------------------------------------------------------------------
// WMMA GEMM:  OUT[M x N] = act(A16[M x K] @ WT16[N x K]^T + bias)
// Block: 256 threads = 8 waves; block tile 32(M) x 256(N); wave tile 16 x 64
// with 4 independent accumulators (4 back-to-back WMMAs per K-step).
// A tile async-copied to LDS; B fragments stream from L2-resident weights.
// grid = (M/32, N/256)
// ---------------------------------------------------------------------------
template <int K>
__global__ __launch_bounds__(256) void wmma_gemm(
    const _Float16* __restrict__ A,   // M x K row-major f16
    const _Float16* __restrict__ WT,  // N x K row-major f16 (transposed weights)
    const float* __restrict__ bias,   // N
    float* __restrict__ out32,        // M x N (f16out == 0)
    _Float16* __restrict__ out16,     // M x N (f16out == 1)
    int N, int relu, int f16out)
{
    __shared__ _Float16 As[32 * K];
    const int tileM = blockIdx.x;
    const int nbase = blockIdx.y * 256;
    const int tid = threadIdx.x;

    // async copy of the contiguous 32 x K activation tile into LDS
    {
        const char* Ag = (const char*)(A + (size_t)tileM * 32 * K);
        const uint32_t lbase = lds_off32(As);
        for (int i = tid; i < (32 * K) / 8; i += 256)
            async_copy_b128(lbase + (uint32_t)i * 16, Ag + (size_t)i * 16);
        async_wait0();
    }
    __syncthreads();

    const int wave = tid >> 5, lane = tid & 31;
    const int lrow = lane & 15, lhal = lane >> 4;
    const int wm = wave >> 2;            // 0..1  M sub-tile
    const int wn = wave & 3;             // 0..3  N sub-column (64 wide)
    const int n0 = nbase + wn * 64;

    const _Float16* Ar = As + (wm * 16 + lrow) * K + lhal * 16;
    const _Float16* Bg = WT + (size_t)(n0 + lrow) * K + lhal * 16;

    v8f c0 = {}, c1 = {}, c2 = {}, c3 = {};
#pragma unroll 2
    for (int k0 = 0; k0 < K; k0 += 32) {
        v16h a  = ldfrag(Ar + k0, Ar + k0 + 8);
        v16h b0 = ldfrag(Bg + k0,              Bg + k0 + 8);
        v16h b1 = ldfrag(Bg + 16 * K + k0,     Bg + 16 * K + k0 + 8);
        v16h b2 = ldfrag(Bg + 32 * K + k0,     Bg + 32 * K + k0 + 8);
        v16h b3 = ldfrag(Bg + 48 * K + k0,     Bg + 48 * K + k0 + 8);
        c0 = wmma_f16(a, b0, c0);
        c1 = wmma_f16(a, b1, c1);
        c2 = wmma_f16(a, b2, c2);
        c3 = wmma_f16(a, b3, c3);
    }

    v8f acc[4] = { c0, c1, c2, c3 };
#pragma unroll
    for (int j = 0; j < 4; ++j) {
        const int n = n0 + j * 16 + lrow;
        const float bb = bias[n];
#pragma unroll
        for (int r = 0; r < 8; ++r) {
            const int m = wm * 16 + lhal * 8 + r;
            const size_t row = (size_t)tileM * 32 + m;
            float v = acc[j][r] + bb;
            if (relu) v = fmaxf(v, 0.f);
            if (f16out) out16[row * N + n] = (_Float16)v;
            else        out32[row * N + n] = v;
        }
    }
}

// ---------------------------------------------------------------------------
// Fused attention for one (batch, head). S=53 padded to 64, DH=32.
// q/k rows async-copied to LDS unguarded: padded rows (and any tail overread,
// which stays inside the workspace) are annihilated by the -1e9 mask.
// ---------------------------------------------------------------------------
__global__ __launch_bounds__(256) void attention_kernel(
    const _Float16* __restrict__ qkv,   // M x 768
    const int* __restrict__ nvalid,
    _Float16* __restrict__ o16)         // M x 256
{
    const int bh = blockIdx.x;
    const int b = bh / NH, h = bh % NH;
    const int tid = threadIdx.x;
    __shared__ _Float16 qs[64 * DH];
    __shared__ _Float16 ks[64 * DH];
    __shared__ _Float16 vt[DH * 64];    // transposed: [d][token]
    __shared__ float    sc[64 * 64];
    __shared__ _Float16 ps[64 * 64];

    // async stage q and k: 64 rows x 32 f16 = 4 b128 chunks per row
    {
        const char* qbase = (const char*)(qkv + ((size_t)b * SEQ) * (3 * DIM) + h * DH);
        const uint32_t qref = lds_off32(qs);
        const uint32_t kref = lds_off32(ks);
        // 256 chunks total -> exactly one per thread
        const int s = tid >> 2, part = tid & 3;
        const char* g = qbase + ((size_t)s * (3 * DIM) + part * 8) * 2;
        const uint32_t loff = (uint32_t)((s * DH + part * 8) * 2);
        async_copy_b128(qref + loff, g);
        async_copy_b128(kref + loff, g + DIM * 2);
    }
    // v staged transposed (synchronous scalar, zero-padded)
    for (int i = tid; i < 64 * DH; i += 256) {
        const int s = i >> 5, d = i & 31;
        _Float16 v = (_Float16)0.f;
        if (s < SEQ)
            v = qkv[((size_t)b * SEQ + s) * (3 * DIM) + 2 * DIM + h * DH + d];
        vt[d * 64 + s] = v;
    }
    async_wait0();
    __syncthreads();

    const int wave = tid >> 5, lane = tid & 31;
    const int lrow = lane & 15, lhal = lane >> 4;
    const float scale = 0.17677669529663687f;  // 1/sqrt(32)
    const int vl = 3 + nvalid[b];

    // scores: 4x4 tiles of 16x16, two per wave (K = 32, single WMMA each)
    for (int t = wave; t < 16; t += 8) {
        const int tm = t >> 2, tn = t & 3;
        v16h a  = ldfrag(qs + (tm * 16 + lrow) * DH + lhal * 16,
                         qs + (tm * 16 + lrow) * DH + lhal * 16 + 8);
        v16h bb = ldfrag(ks + (tn * 16 + lrow) * DH + lhal * 16,
                         ks + (tn * 16 + lrow) * DH + lhal * 16 + 8);
        v8f c = {};
        c = wmma_f16(a, bb, c);
#pragma unroll
        for (int r = 0; r < 8; ++r) {
            const int m = tm * 16 + lhal * 8 + r;
            const int n = tn * 16 + lrow;
            sc[m * 64 + n] = c[r] * scale + ((n >= vl) ? -1.0e9f : 0.f);
        }
    }
    __syncthreads();

    // masked softmax: one thread per row
    if (tid < 64) {
        float mx = -1.0e30f;
        for (int j = 0; j < 64; ++j) mx = fmaxf(mx, sc[tid * 64 + j]);
        float sum = 0.f;
        for (int j = 0; j < 64; ++j) {
            float e = __expf(sc[tid * 64 + j] - mx);
            sc[tid * 64 + j] = e;
            sum += e;
        }
        const float inv = 1.f / sum;
        for (int j = 0; j < 64; ++j) ps[tid * 64 + j] = (_Float16)(sc[tid * 64 + j] * inv);
    }
    __syncthreads();

    // o = p(64x64) @ v(64x32): 8 tiles -> one per wave, K = 64 (2 WMMAs)
    {
        const int tm = wave >> 1, tn = wave & 1;
        v8f c = {};
#pragma unroll
        for (int k0 = 0; k0 < 64; k0 += 32) {
            v16h a  = ldfrag(ps + (tm * 16 + lrow) * 64 + k0 + lhal * 16,
                             ps + (tm * 16 + lrow) * 64 + k0 + lhal * 16 + 8);
            v16h bb = ldfrag(vt + (tn * 16 + lrow) * 64 + k0 + lhal * 16,
                             vt + (tn * 16 + lrow) * 64 + k0 + lhal * 16 + 8);
            c = wmma_f16(a, bb, c);
        }
#pragma unroll
        for (int r = 0; r < 8; ++r) {
            const int m = tm * 16 + lhal * 8 + r;   // token
            const int n = tn * 16 + lrow;           // dh
            if (m < SEQ)
                o16[((size_t)b * SEQ + m) * DIM + h * DH + n] = (_Float16)c[r];
        }
    }
}

// ---------------------------------------------------------------------------
// x = LN(x + t) ; writes fp32 residual stream and f16 GEMM feed.
// ---------------------------------------------------------------------------
__global__ __launch_bounds__(256) void add_ln_kernel(
    const float* __restrict__ xin, const float* __restrict__ add,
    const float* __restrict__ g, const float* __restrict__ bta,
    float* __restrict__ xout, _Float16* __restrict__ x16out)
{
    const size_t row = blockIdx.x;
    const int tid = threadIdx.x;
    const int wave = tid >> 5, lane = tid & 31;
    float v = xin[row * DIM + tid] + add[row * DIM + tid];
    float s = v, q = v * v;
#pragma unroll
    for (int off = 16; off; off >>= 1) {
        s += __shfl_down(s, off, 32);
        q += __shfl_down(q, off, 32);
    }
    __shared__ float rs[8], rq[8];
    __shared__ float mean_s, inv_s;
    if (lane == 0) { rs[wave] = s; rq[wave] = q; }
    __syncthreads();
    if (tid == 0) {
        float S = 0.f, Q = 0.f;
        for (int i = 0; i < 8; ++i) { S += rs[i]; Q += rq[i]; }
        float m = S * (1.f / DIM);
        float var = Q * (1.f / DIM) - m * m;
        mean_s = m;
        inv_s = rsqrtf(var + 1e-5f);
    }
    __syncthreads();
    float y = (v - mean_s) * inv_s * g[tid] + bta[tid];
    xout[row * DIM + tid] = y;
    x16out[row * DIM + tid] = (_Float16)y;
}

// ---------------------------------------------------------------------------
__global__ void gather_cls(const _Float16* __restrict__ x16, _Float16* __restrict__ cls16) {
    const int i = blockIdx.x * 256 + threadIdx.x;
    const int b = i >> 8, d = i & 255;
    cls16[i] = x16[((size_t)b * SEQ) * DIM + d];
}

// ---------------------------------------------------------------------------
// Final heads: action_mean = tanh(cls @ projW + projb), value = h2 @ W3 + b3
// ---------------------------------------------------------------------------
__global__ __launch_bounds__(256) void head_kernel(
    const float* __restrict__ x32, const _Float16* __restrict__ h2,
    const float* __restrict__ projW, const float* __restrict__ projb,
    const float* __restrict__ vhW3, const float* __restrict__ vhb3,
    float* __restrict__ out)
{
    const int b = blockIdx.x, tid = threadIdx.x;
    const int wave = tid >> 5, lane = tid & 31;
    const float c = x32[((size_t)b * SEQ) * DIM + tid];
    float a0 = c * projW[tid * 2 + 0];
    float a1 = c * projW[tid * 2 + 1];
    float a2 = (float)h2[(size_t)b * DIM + tid] * vhW3[tid];
#pragma unroll
    for (int off = 16; off; off >>= 1) {
        a0 += __shfl_down(a0, off, 32);
        a1 += __shfl_down(a1, off, 32);
        a2 += __shfl_down(a2, off, 32);
    }
    __shared__ float r0[8], r1[8], r2[8];
    if (lane == 0) { r0[wave] = a0; r1[wave] = a1; r2[wave] = a2; }
    __syncthreads();
    if (tid == 0) {
        float s0 = 0.f, s1 = 0.f, s2 = 0.f;
        for (int i = 0; i < 8; ++i) { s0 += r0[i]; s1 += r1[i]; s2 += r2[i]; }
        out[b * 3 + 0] = tanhf(s0 + projb[0]);
        out[b * 3 + 1] = tanhf(s1 + projb[1]);
        out[b * 3 + 2] = s2 + vhb3[0];
    }
}

// ---------------------------------------------------------------------------
extern "C" void kernel_launch(void* const* d_in, const int* in_sizes, int n_in,
                              void* d_out, int out_size, void* d_ws, size_t ws_size,
                              hipStream_t stream) {
    (void)in_sizes; (void)n_in; (void)out_size; (void)ws_size;
    const float* obs      = (const float*)d_in[0];
    const float* cls_emb  = (const float*)d_in[1];
    const float* type_cls = (const float*)d_in[2];
    const float* type_ctx = (const float*)d_in[3];
    const float* type_pred= (const float*)d_in[4];
    const float* type_boid= (const float*)d_in[5];
    const float* ctxW     = (const float*)d_in[6];
    const float* ctxb     = (const float*)d_in[7];
    const float* predW    = (const float*)d_in[8];
    const float* predb    = (const float*)d_in[9];
    const float* boidW    = (const float*)d_in[10];
    const float* boidb    = (const float*)d_in[11];
    const float* qkvW     = (const float*)d_in[12];
    const float* qkvb     = (const float*)d_in[13];
    const float* outW     = (const float*)d_in[14];
    const float* outb     = (const float*)d_in[15];
    const float* ln1g     = (const float*)d_in[16];
    const float* ln1b     = (const float*)d_in[17];
    const float* fW1      = (const float*)d_in[18];
    const float* fb1      = (const float*)d_in[19];
    const float* fW2      = (const float*)d_in[20];
    const float* fb2      = (const float*)d_in[21];
    const float* ln2g     = (const float*)d_in[22];
    const float* ln2b     = (const float*)d_in[23];
    const float* projW    = (const float*)d_in[24];
    const float* projb    = (const float*)d_in[25];
    const float* vhW1     = (const float*)d_in[26];
    const float* vhb1     = (const float*)d_in[27];
    const float* vhW2     = (const float*)d_in[28];
    const float* vhb2     = (const float*)d_in[29];
    const float* vhW3     = (const float*)d_in[30];
    const float* vhb3     = (const float*)d_in[31];
    float* out = (float*)d_out;

    char* ws = (char*)d_ws;
    size_t off = 0;
    auto take = [&](size_t bytes) -> char* {
        char* p = ws + off;
        off = (off + bytes + 255) & ~(size_t)255;
        return p;
    };
    float*    x32   = (float*)   take(MTOT * DIM * 4);
    _Float16* x16   = (_Float16*)take(MTOT * DIM * 2);
    _Float16* qkv16 = (_Float16*)take(MTOT * 3 * DIM * 2);
    _Float16* o16   = (_Float16*)take(MTOT * DIM * 2);
    _Float16* hh16  = (_Float16*)take(MTOT * FF * 2);
    float*    t32   = (float*)   take(MTOT * DIM * 4);
    _Float16* wqkvT = (_Float16*)take((size_t)NL * 3 * DIM * DIM * 2);
    _Float16* woutT = (_Float16*)take((size_t)NL * DIM * DIM * 2);
    _Float16* w1T   = (_Float16*)take((size_t)NL * FF * DIM * 2);
    _Float16* w2T   = (_Float16*)take((size_t)NL * DIM * FF * 2);
    _Float16* vh1T  = (_Float16*)take((size_t)DIM * DIM * 2);
    _Float16* vh2T  = (_Float16*)take((size_t)DIM * DIM * 2);
    int*      nval  = (int*)     take((size_t)BN * 4);
    _Float16* cls16 = (_Float16*)take((size_t)BN * DIM * 2);
    _Float16* h1    = (_Float16*)take((size_t)BN * DIM * 2);
    _Float16* h2    = (_Float16*)take((size_t)BN * DIM * 2);

    auto cgrid = [](size_t n) { return dim3((unsigned)((n + 255) / 256)); };

    // ---- weight conversion ----
    for (int l = 0; l < NL; ++l) {
        convert_wT<<<cgrid((size_t)DIM * 3 * DIM), 256, 0, stream>>>(
            qkvW + (size_t)l * DIM * 3 * DIM, wqkvT + (size_t)l * 3 * DIM * DIM, DIM, 3 * DIM);
        convert_wT<<<cgrid((size_t)DIM * DIM), 256, 0, stream>>>(
            outW + (size_t)l * DIM * DIM, woutT + (size_t)l * DIM * DIM, DIM, DIM);
        convert_wT<<<cgrid((size_t)DIM * FF), 256, 0, stream>>>(
            fW1 + (size_t)l * DIM * FF, w1T + (size_t)l * FF * DIM, DIM, FF);
        convert_wT<<<cgrid((size_t)FF * DIM), 256, 0, stream>>>(
            fW2 + (size_t)l * FF * DIM, w2T + (size_t)l * DIM * FF, FF, DIM);
    }
    convert_wT<<<cgrid((size_t)DIM * DIM), 256, 0, stream>>>(vhW1, vh1T, DIM, DIM);
    convert_wT<<<cgrid((size_t)DIM * DIM), 256, 0, stream>>>(vhW2, vh2T, DIM, DIM);

    // ---- embedding / packing ----
    embed_kernel<<<BN, 256, 0, stream>>>(obs, cls_emb, type_cls, type_ctx, type_pred,
                                         type_boid, ctxW, ctxb, predW, predb, boidW, boidb,
                                         x32, x16, nval);

    const unsigned mt32 = (unsigned)(MTOT / 32);

    // ---- transformer layers ----
    for (int l = 0; l < NL; ++l) {
        wmma_gemm<DIM><<<dim3(mt32, 3), 256, 0, stream>>>(
            x16, wqkvT + (size_t)l * 3 * DIM * DIM, qkvb + (size_t)l * 3 * DIM,
            nullptr, qkv16, 3 * DIM, 0, 1);

        attention_kernel<<<BN * NH, 256, 0, stream>>>(qkv16, nval, o16);

        wmma_gemm<DIM><<<dim3(mt32, 1), 256, 0, stream>>>(
            o16, woutT + (size_t)l * DIM * DIM, outb + (size_t)l * DIM,
            t32, nullptr, DIM, 0, 0);

        add_ln_kernel<<<(unsigned)MTOT, 256, 0, stream>>>(
            x32, t32, ln1g + (size_t)l * DIM, ln1b + (size_t)l * DIM, x32, x16);

        wmma_gemm<DIM><<<dim3(mt32, 4), 256, 0, stream>>>(
            x16, w1T + (size_t)l * FF * DIM, fb1 + (size_t)l * FF,
            nullptr, hh16, FF, 1, 1);

        wmma_gemm<FF><<<dim3(mt32, 1), 256, 0, stream>>>(
            hh16, w2T + (size_t)l * DIM * FF, fb2 + (size_t)l * DIM,
            t32, nullptr, DIM, 0, 0);

        add_ln_kernel<<<(unsigned)MTOT, 256, 0, stream>>>(
            x32, t32, ln2g + (size_t)l * DIM, ln2b + (size_t)l * DIM, x32, x16);
    }

    // ---- heads ----
    gather_cls<<<BN, 256, 0, stream>>>(x16, cls16);
    wmma_gemm<DIM><<<dim3(BN / 32, 1), 256, 0, stream>>>(cls16, vh1T, vhb1, nullptr, h1, DIM, 1, 1);
    wmma_gemm<DIM><<<dim3(BN / 32, 1), 256, 0, stream>>>(h1, vh2T, vhb2, nullptr, h2, DIM, 1, 1);
    head_kernel<<<BN, 256, 0, stream>>>(x32, h2, projW, projb, vhW3, vhb3, out);
}
